// EdgeEarlyInteraction2_76519137345680
// MI455X (gfx1250) — matrix operand order repaired and checked by reading, not compiled, not run on hardware
//
#include <hip/hip_runtime.h>
#include <hip/hip_fp16.h>

// ---------------- problem constants (match reference) ----------------
#define BB     256      // graph pairs
#define NQv    96
#define NCv    120
#define EQv    192
#define ECv    240
#define MPAD   256
#define RAWv   32
#define HDIM   64
#define MSGD   128
#define SDIMv  64
#define EPG    (EQv + ECv)          // 432 edges per pair
#define NNODES (BB * (NQv + NCv))   // 55296  (= 432*128)
#define NEDGES (BB * EPG)           // 110592 (= 864*128)

typedef __attribute__((ext_vector_type(16))) _Float16 v16h;
typedef __attribute__((ext_vector_type(8)))  _Float16 v8h;
typedef __attribute__((ext_vector_type(8)))  float    v8f;

// 16-bit fragment layouts used here (contiguous k order, same for A and B):
//  A (16x32): lane = m + 16*half holds row m; its 16 halves are two
//    contiguous 8-elem chunks at k-offsets half*16 and half*16+8
//    -> direct 2x16B loads from row-major f16.
//  B (32x16): lane = n + 16*half holds k = half*16 + e. We pre-swizzle B
//    operands into "fragment order" so each lane loads one contiguous 32B
//    v16h:  dst[(kchunk*(N/16)+ctile)*512 + lane*16 + e].

__device__ __forceinline__ v16h frag_from_chunks(v8h lo, v8h hi) {
    v16h r;
    #pragma unroll
    for (int i = 0; i < 8; ++i) { r[i] = lo[i]; r[8 + i] = hi[i]; }
    return r;
}

// =====================================================================
// Generic WMMA GEMM over a "virtual" X = concat(src0|src1|src2), all f16,
// with optional per-row gather indices; weights pre-swizzled f16.
// Block: 256 threads = 8 waves; block tile = 128 rows x 32 cols.
// NO LDS, NO barriers: A and B fragments load directly from global.
// Assumes M%128==0, segment widths %32==0, N%32==0.
// mode: 0=f16 store, 1=f16 add, 2=f32 atomic scatter (row->oridx[row]).
// =====================================================================
__global__ __launch_bounds__(256)
void gemm_rv(const _Float16* __restrict__ s0, int c0, const int* __restrict__ r0,
             const _Float16* __restrict__ s1, int c1, const int* __restrict__ r1,
             const _Float16* __restrict__ s2, int c2, const int* __restrict__ r2,
             const _Float16* __restrict__ Wsw, const float* __restrict__ bias,
             void* __restrict__ out, const int* __restrict__ oridx,
             int N, int relu, int mode, int maskMod, int maskLimit)
{
    const int t    = threadIdx.x;
    const int wave = t >> 5, lane = t & 31;
    const int m16  = lane & 15, half = lane >> 4;
    const int blockRow = blockIdx.x * 128;
    const int colBase  = blockIdx.y * 32;
    const int arow = blockRow + wave * 16 + m16;   // A row owned by this lane
    const size_t wstep = (size_t)(N >> 4) * 512;
    const _Float16* wp = Wsw + (size_t)(colBase >> 4) * 512 + lane * 16;

    v8f acc0 = {}, acc1 = {};
    for (int seg = 0; seg < 3; ++seg) {
        const _Float16* sp; int scol; const int* ridx;
        if (seg == 0)      { sp = s0; scol = c0; ridx = r0; }
        else if (seg == 1) { sp = s1; scol = c1; ridx = r1; }
        else               { sp = s2; scol = c2; ridx = r2; }
        if (scol == 0) continue;
        const int rr = ridx ? ridx[arow] : arow;
        const _Float16* rp = sp + (size_t)rr * scol + half * 16;

        for (int kk = 0; kk < scol; kk += 32) {
            v16h af  = frag_from_chunks(*(const v8h*)rp, *(const v8h*)(rp + 8));
            v16h bf0 = *(const v16h*)wp;
            v16h bf1 = *(const v16h*)(wp + 512);
            if (kk + 32 < scol) __builtin_prefetch(rp + 32, 0, 3);
            acc0 = __builtin_amdgcn_wmma_f32_16x16x32_f16(false, af, false, bf0,
                                                          (short)0, acc0, false, false);
            acc1 = __builtin_amdgcn_wmma_f32_16x16x32_f16(false, af, false, bf1,
                                                          (short)0, acc1, false, false);
            rp += 32;
            wp += wstep;
        }
    }
    // ---- epilogue: C layout VGPR r -> M = r + 8*half, N = lane&15 ----
    const int nA = colBase + m16, nB = nA + 16;
    const float bvA = bias ? bias[nA] : 0.f;
    const float bvB = bias ? bias[nB] : 0.f;
    #pragma unroll
    for (int r = 0; r < 8; ++r) {
        int orow = blockRow + wave * 16 + r + 8 * half;
        float v0 = acc0[r] + bvA;
        float v1 = acc1[r] + bvB;
        if (relu) { v0 = fmaxf(v0, 0.f); v1 = fmaxf(v1, 0.f); }
        if (maskMod > 0 && (orow % maskMod) >= maskLimit) { v0 = 0.f; v1 = 0.f; }
        int drow = oridx ? oridx[orow] : orow;
        if (mode == 2) {
            float* dst = (float*)out + (size_t)drow * N;
            atomicAdd(dst + nA, v0);
            atomicAdd(dst + nB, v1);
        } else {
            _Float16* dst = (_Float16*)out + (size_t)drow * N;
            if (mode == 0) { dst[nA] = (_Float16)v0; dst[nB] = (_Float16)v1; }
            else {
                dst[nA] = (_Float16)((float)dst[nA] + v0);
                dst[nB] = (_Float16)((float)dst[nB] + v1);
            }
        }
    }
}

// =====================================================================
// Batched WMMA GEMM, M fixed = 256 rows/batch; A f16 row-major (direct
// fragment loads); B pre-swizzled per batch; C f32 or f16.  No LDS.
// C[b, row<rowLimit, n] = alpha * sum_k A(b,m,k) * Bsw(b,k,n)
// =====================================================================
__global__ __launch_bounds__(256)
void gemm_bat(const _Float16* __restrict__ A, int lda, long long sA,
              const _Float16* __restrict__ Bsw, long long sB,
              void* __restrict__ C, int cHalf, int ldc, long long sC,
              int N, int K, float alpha, int rowLimit)
{
    const int t = threadIdx.x, wave = t >> 5, lane = t & 31;
    const int m16 = lane & 15, half = lane >> 4;
    const int blockRow = blockIdx.x * 128;
    const int colBase  = blockIdx.y * 32;
    const _Float16* rp = A + (size_t)((long long)blockIdx.z * sA)
                           + (size_t)(blockRow + wave * 16 + m16) * lda + half * 16;
    const size_t wstep = (size_t)(N >> 4) * 512;
    const _Float16* wp = Bsw + (size_t)((long long)blockIdx.z * sB)
                             + (size_t)(colBase >> 4) * 512 + lane * 16;

    v8f acc0 = {}, acc1 = {};
    for (int k0 = 0; k0 < K; k0 += 32) {
        v16h af  = frag_from_chunks(*(const v8h*)rp, *(const v8h*)(rp + 8));
        v16h bf0 = *(const v16h*)wp;
        v16h bf1 = *(const v16h*)(wp + 512);
        acc0 = __builtin_amdgcn_wmma_f32_16x16x32_f16(false, af, false, bf0,
                                                      (short)0, acc0, false, false);
        acc1 = __builtin_amdgcn_wmma_f32_16x16x32_f16(false, af, false, bf1,
                                                      (short)0, acc1, false, false);
        rp += 32;
        wp += wstep;
    }
    const int nA = colBase + m16, nB = nA + 16;
    #pragma unroll
    for (int r = 0; r < 8; ++r) {
        int orow = blockRow + wave * 16 + r + 8 * half;
        if (orow < rowLimit) {
            float v0 = alpha * acc0[r], v1 = alpha * acc1[r];
            if (cHalf) {
                _Float16* Cp = (_Float16*)C + (size_t)((long long)blockIdx.z * sC);
                Cp[(size_t)orow * ldc + nA] = (_Float16)v0;
                Cp[(size_t)orow * ldc + nB] = (_Float16)v1;
            } else {
                float* Cp = (float*)C + (size_t)((long long)blockIdx.z * sC);
                Cp[(size_t)orow * ldc + nA] = v0;
                Cp[(size_t)orow * ldc + nB] = v1;
            }
        }
    }
}

// ------------------ swizzle / utility kernels ------------------
// f32 W[K,N] row-major -> frag-ordered f16
__global__ void swizW_k(const float* __restrict__ W, _Float16* __restrict__ dst,
                        int K, int N) {
    int i = blockIdx.x * 256 + threadIdx.x;
    if (i >= K * N) return;
    int k = i / N, n = i % N;
    int kin = k & 31, halfk = kin >> 4, e = kin & 15;
    int lane = (n & 15) + 16 * halfk;
    size_t o = ((size_t)((k >> 5) * (N >> 4) + (n >> 4))) * 512 + lane * 16 + e;
    dst[o] = (_Float16)W[(size_t)k * N + n];
}

// f16 batched B [b][K,N] (or [b][N,K] if trans) -> frag-ordered per batch
__global__ void swizB_k(const _Float16* __restrict__ src, _Float16* __restrict__ dst,
                        int K, int N, int trans, int lds, long long sSrc) {
    int i = blockIdx.x * 256 + threadIdx.x;
    if (i >= K * N) return;
    int k = i / N, n = i % N;
    const _Float16* s = src + (size_t)((long long)blockIdx.z * sSrc);
    _Float16 v = trans ? s[(size_t)n * lds + k] : s[(size_t)k * lds + n];
    int kin = k & 31, halfk = kin >> 4, e = kin & 15;
    int lane = (n & 15) + 16 * halfk;
    size_t o = (size_t)blockIdx.z * ((size_t)K * N)
             + ((size_t)((k >> 5) * (N >> 4) + (n >> 4))) * 512 + lane * 16 + e;
    dst[o] = v;
}

__global__ void fill0_k(float* __restrict__ p, size_t n) {
    size_t i = (size_t)blockIdx.x * 256 + threadIdx.x;
    if (i < n) p[i] = 0.f;
}

__global__ void cast_h_k(const float* __restrict__ in, _Float16* __restrict__ out,
                         size_t n) {
    size_t i = (size_t)blockIdx.x * 256 + threadIdx.x;
    if (i < n) out[i] = (_Float16)in[i];
}

// split e [NEDGES,128] f16 -> padded qs/cs [B,256,128] f16
__global__ void split_pad_k(const _Float16* __restrict__ e,
                            _Float16* __restrict__ qs, _Float16* __restrict__ cs) {
    size_t i = (size_t)blockIdx.x * 256 + threadIdx.x;
    if (i >= (size_t)BB * MPAD * MSGD) return;
    int d = (int)(i & (MSGD - 1));
    int r = (int)((i >> 7) & (MPAD - 1));
    int b = (int)(i >> 15);
    _Float16 z = (_Float16)0.f;
    qs[i] = (r < EQv) ? e[((size_t)b * EPG + r) * MSGD + d] : z;
    cs[i] = (r < ECv) ? e[((size_t)b * EPG + EQv + r) * MSGD + d] : z;
}

// logsumexp normalization over the last axis (c): one block per (b,q)
__global__ __launch_bounds__(256) void sink_rownorm_k(float* __restrict__ la) {
    __shared__ float red[256];
    int t = threadIdx.x;
    size_t base = (size_t)blockIdx.x * 256;
    float x = la[base + t];
    red[t] = x; __syncthreads();
    for (int s = 128; s > 0; s >>= 1) { if (t < s) red[t] = fmaxf(red[t], red[t + s]); __syncthreads(); }
    float mx = red[0]; __syncthreads();
    red[t] = __expf(x - mx); __syncthreads();
    for (int s = 128; s > 0; s >>= 1) { if (t < s) red[t] += red[t + s]; __syncthreads(); }
    la[base + t] = x - (mx + __logf(red[0]));
}

// logsumexp normalization over the q axis: one block per (b,c).
// Final pass (do_exp) emits the f16 transport plan in both layouts.
__global__ __launch_bounds__(256)
void sink_colnorm_k(float* __restrict__ la, _Float16* __restrict__ plan,
                    _Float16* __restrict__ planT, int do_exp) {
    __shared__ float red[256];
    int t = threadIdx.x;
    int b = blockIdx.x >> 8, c = blockIdx.x & 255;
    size_t idx = (size_t)b * 65536 + (size_t)t * 256 + c;
    float x = la[idx];
    red[t] = x; __syncthreads();
    for (int s = 128; s > 0; s >>= 1) { if (t < s) red[t] = fmaxf(red[t], red[t + s]); __syncthreads(); }
    float mx = red[0]; __syncthreads();
    red[t] = __expf(x - mx); __syncthreads();
    for (int s = 128; s > 0; s >>= 1) { if (t < s) red[t] += red[t + s]; __syncthreads(); }
    float v = x - (mx + __logf(red[0]));
    if (do_exp) {
        float pv = __expf(v);
        plan [(size_t)b * 65536 + (size_t)t * 256 + c] = (_Float16)pv;  // [b,q,c]
        planT[(size_t)b * 65536 + (size_t)c * 256 + t] = (_Float16)pv;  // [b,c,q]
    } else {
        la[idx] = v;
    }
}

// score[b] = -sum |qs - tb| over 256x128 (qs f16, tb f32)
__global__ __launch_bounds__(256) void score_k(const _Float16* __restrict__ qs,
                                               const float* __restrict__ tb,
                                               float* __restrict__ out) {
    __shared__ float red[256];
    int b = blockIdx.x, t = threadIdx.x;
    size_t base = (size_t)b * MPAD * MSGD;
    float s = 0.f;
    for (int i = t; i < MPAD * MSGD; i += 256)
        s += fabsf((float)qs[base + i] - tb[base + i]);
    red[t] = s; __syncthreads();
    for (int st = 128; st > 0; st >>= 1) { if (t < st) red[t] += red[t + st]; __syncthreads(); }
    if (t == 0) out[b] = -red[0];
}

// ------------------ host-side launch helpers ------------------
static inline void gemm(hipStream_t st,
                        const _Float16* s0, int c0, const int* r0,
                        const _Float16* s1, int c1, const int* r1,
                        const _Float16* s2, int c2, const int* r2,
                        const _Float16* Wsw, const float* bias,
                        void* out, const int* oridx,
                        int M, int N, int relu, int mode,
                        int maskMod = 0, int maskLimit = 0)
{
    dim3 g(M / 128, N / 32);
    gemm_rv<<<g, 256, 0, st>>>(s0, c0, r0, s1, c1, r1, s2, c2, r2,
                               Wsw, bias, out, oridx, N, relu, mode,
                               maskMod, maskLimit);
}

static inline void fill0(hipStream_t st, float* p, size_t n) {
    fill0_k<<<(unsigned)((n + 255) / 256), 256, 0, st>>>(p, n);
}

static inline void castH(hipStream_t st, const float* in, _Float16* out, size_t n) {
    cast_h_k<<<(unsigned)((n + 255) / 256), 256, 0, st>>>(in, out, n);
}

static inline void swizW(hipStream_t st, const float* W, _Float16* dst, int K, int N) {
    swizW_k<<<(K * N + 255) / 256, 256, 0, st>>>(W, dst, K, N);
}

static inline void swizB(hipStream_t st, const _Float16* src, _Float16* dst,
                         int K, int N, int trans, int lds, long long sSrc) {
    dim3 g((K * N + 255) / 256, 1, BB);
    swizB_k<<<g, 256, 0, st>>>(src, dst, K, N, trans, lds, sSrc);
}

extern "C" void kernel_launch(void* const* d_in, const int* in_sizes, int n_in,
                              void* d_out, int out_size, void* d_ws, size_t ws_size,
                              hipStream_t stream) {
    (void)in_sizes; (void)n_in; (void)out_size; (void)ws_size;
    const float* node_f = (const float*)d_in[0];
    const float* edge_f = (const float*)d_in[1];
    const int*   fidx   = (const int*)d_in[2];
    const int*   tidx   = (const int*)d_in[3];
    const float* encNW  = (const float*)d_in[4];
    const float* encNb  = (const float*)d_in[5];
    const float* encEW  = (const float*)d_in[6];
    const float* encEb  = (const float*)d_in[7];
    const float* mW1 = (const float*)d_in[8];  const float* mb1 = (const float*)d_in[9];
    const float* mW2 = (const float*)d_in[10]; const float* mb2 = (const float*)d_in[11];
    const float* uW1 = (const float*)d_in[12]; const float* ub1 = (const float*)d_in[13];
    const float* uW2 = (const float*)d_in[14]; const float* ub2 = (const float*)d_in[15];
    const float* iW1 = (const float*)d_in[16]; const float* ib1 = (const float*)d_in[17];
    const float* iW2 = (const float*)d_in[18]; const float* ib2 = (const float*)d_in[19];
    const float* sW1 = (const float*)d_in[20]; const float* sb1 = (const float*)d_in[21];
    const float* sW2 = (const float*)d_in[22]; const float* sb2 = (const float*)d_in[23];

    // ---- workspace carve-out (256B aligned slices) ----
    char* base = (char*)d_ws;
    size_t off = 0;
    auto carve = [&](size_t bytes) -> void* {
        void* r = base + off;
        off += (bytes + 255) & ~(size_t)255;
        return r;
    };
    _Float16* node_h = (_Float16*)carve((size_t)NNODES * RAWv * 2);
    _Float16* edge_h = (_Float16*)carve((size_t)NEDGES * RAWv * 2);
    _Float16* h0    = (_Float16*)carve((size_t)NNODES * HDIM * 2);
    _Float16* h     = (_Float16*)carve((size_t)NNODES * HDIM * 2);
    _Float16* e0    = (_Float16*)carve((size_t)NEDGES * MSGD * 2);
    _Float16* e     = (_Float16*)carve((size_t)NEDGES * MSGD * 2);
    _Float16* inter = (_Float16*)carve((size_t)NEDGES * MSGD * 2);
    _Float16* comb  = (_Float16*)carve((size_t)NEDGES * MSGD * 2);
    _Float16* hid   = (_Float16*)carve((size_t)NEDGES * 256 * 2);
    float*    aggf  = (float*)carve((size_t)NNODES * MSGD * 4);
    float*    aggb  = (float*)carve((size_t)NNODES * MSGD * 4);
    _Float16* aggfh = (_Float16*)carve((size_t)NNODES * MSGD * 2);
    _Float16* aggbh = (_Float16*)carve((size_t)NNODES * MSGD * 2);
    _Float16* qs    = (_Float16*)carve((size_t)BB * MPAD * MSGD * 2);
    _Float16* cs    = (_Float16*)carve((size_t)BB * MPAD * MSGD * 2);
    _Float16* fq    = (_Float16*)carve((size_t)BB * MPAD * SDIMv * 2);
    _Float16* fc    = (_Float16*)carve((size_t)BB * MPAD * SDIMv * 2);
    float*    la    = (float*)carve((size_t)BB * MPAD * MPAD * 4);  // f32 logits
    _Float16* plan  = (_Float16*)carve((size_t)BB * MPAD * MPAD * 2);
    _Float16* planT = (_Float16*)carve((size_t)BB * MPAD * MPAD * 2);
    float*    tb    = (float*)carve((size_t)BB * MPAD * MSGD * 4);
    // swizzled weights (frag-ordered f16)
    _Float16* wEncN = (_Float16*)carve((size_t)RAWv * HDIM * 2);
    _Float16* wEncE = (_Float16*)carve((size_t)RAWv * MSGD * 2);
    _Float16* wM1   = (_Float16*)carve((size_t)256 * 256 * 2);
    _Float16* wM2   = (_Float16*)carve((size_t)256 * 128 * 2);
    _Float16* wU1   = (_Float16*)carve((size_t)320 * 128 * 2);
    _Float16* wU2   = (_Float16*)carve((size_t)128 * 64 * 2);
    _Float16* wI1   = (_Float16*)carve((size_t)256 * 256 * 2);
    _Float16* wI2   = (_Float16*)carve((size_t)256 * 128 * 2);
    _Float16* wS1   = (_Float16*)carve((size_t)128 * 64 * 2);
    _Float16* wS2   = (_Float16*)carve((size_t)64 * 64 * 2);
    // swizzled batched-B scratch
    _Float16* csw = (_Float16*)carve((size_t)BB * 256 * 128 * 2);
    _Float16* qsw = (_Float16*)carve((size_t)BB * 256 * 128 * 2);
    _Float16* fsw = (_Float16*)carve((size_t)BB * 64 * 256 * 2);

    const size_t splitGrid = ((size_t)BB * MPAD * MSGD + 255) / 256;

    // ---- one-time weight swizzles + input casts ----
    swizW(stream, encNW, wEncN, RAWv, HDIM);
    swizW(stream, encEW, wEncE, RAWv, MSGD);
    swizW(stream, mW1, wM1, 256, 256);
    swizW(stream, mW2, wM2, 256, 128);
    swizW(stream, uW1, wU1, 320, 128);
    swizW(stream, uW2, wU2, 128, 64);
    swizW(stream, iW1, wI1, 256, 256);
    swizW(stream, iW2, wI2, 256, 128);
    swizW(stream, sW1, wS1, 128, 64);
    swizW(stream, sW2, wS2, 64, 64);
    castH(stream, node_f, node_h, (size_t)NNODES * RAWv);
    castH(stream, edge_f, edge_h, (size_t)NEDGES * RAWv);

    // ---- graph encoders ----
    gemm(stream, node_h, RAWv, nullptr, nullptr, 0, nullptr, nullptr, 0, nullptr,
         wEncN, encNb, h0, nullptr, NNODES, HDIM, 0, 0);
    gemm(stream, edge_h, RAWv, nullptr, nullptr, 0, nullptr, nullptr, 0, nullptr,
         wEncE, encEb, e0, nullptr, NEDGES, MSGD, 0, 0);

    for (int ts = 0; ts < 2; ++ts) {
        hipMemcpyAsync(h, h0, (size_t)NNODES * HDIM * sizeof(_Float16),
                       hipMemcpyDeviceToDevice, stream);
        hipMemcpyAsync(e, e0, (size_t)NEDGES * MSGD * sizeof(_Float16),
                       hipMemcpyDeviceToDevice, stream);
        fill0(stream, (float*)inter, (size_t)NEDGES * MSGD / 2);   // f16 zeros

        for (int ps = 0; ps < 3; ++ps) {
            // interaction encoder: comb = MLP2([e, inter])
            gemm(stream, e, MSGD, nullptr, inter, MSGD, nullptr, nullptr, 0, nullptr,
                 wI1, ib1, hid, nullptr, NEDGES, 256, 1, 0);
            gemm(stream, hid, 256, nullptr, nullptr, 0, nullptr, nullptr, 0, nullptr,
                 wI2, ib2, comb, nullptr, NEDGES, MSGD, 0, 0);

            // bidirectional messages; segment-sum fused into GEMM2 (f32 atomics)
            fill0(stream, aggf, (size_t)NNODES * MSGD);
            fill0(stream, aggb, (size_t)NNODES * MSGD);
            gemm(stream, h, HDIM, fidx, h, HDIM, tidx, comb, MSGD, nullptr,
                 wM1, mb1, hid, nullptr, NEDGES, 256, 1, 0);
            gemm(stream, hid, 256, nullptr, nullptr, 0, nullptr, nullptr, 0, nullptr,
                 wM2, mb2, aggf, tidx, NEDGES, MSGD, 0, 2);
            gemm(stream, h, HDIM, tidx, h, HDIM, fidx, comb, MSGD, nullptr,
                 wM1, mb1, hid, nullptr, NEDGES, 256, 1, 0);
            gemm(stream, hid, 256, nullptr, nullptr, 0, nullptr, nullptr, 0, nullptr,
                 wM2, mb2, aggb, fidx, NEDGES, MSGD, 0, 2);
            castH(stream, aggf, aggfh, (size_t)NNODES * MSGD);
            castH(stream, aggb, aggbh, (size_t)NNODES * MSGD);

            // node update: h = MLP2([h, agg_f, agg_b])
            gemm(stream, h, HDIM, nullptr, aggfh, MSGD, nullptr, aggbh, MSGD, nullptr,
                 wU1, ub1, hid, nullptr, NNODES, 128, 1, 0);
            gemm(stream, hid, 128, nullptr, nullptr, 0, nullptr, nullptr, 0, nullptr,
                 wU2, ub2, h, nullptr, NNODES, HDIM, 0, 0);

            // propagation messages: e = mf2 + mb2 (second pass accumulates)
            gemm(stream, h, HDIM, fidx, h, HDIM, tidx, comb, MSGD, nullptr,
                 wM1, mb1, hid, nullptr, NEDGES, 256, 1, 0);
            gemm(stream, hid, 256, nullptr, nullptr, 0, nullptr, nullptr, 0, nullptr,
                 wM2, mb2, e, nullptr, NEDGES, MSGD, 0, 0);
            gemm(stream, h, HDIM, tidx, h, HDIM, fidx, comb, MSGD, nullptr,
                 wM1, mb1, hid, nullptr, NEDGES, 256, 1, 0);
            gemm(stream, hid, 256, nullptr, nullptr, 0, nullptr, nullptr, 0, nullptr,
                 wM2, mb2, e, nullptr, NEDGES, MSGD, 0, 1);

            // interaction features from transport plan (plan exists in ts==1)
            if (ts == 1 && ps < 2) {
                split_pad_k<<<(unsigned)splitGrid, 256, 0, stream>>>(e, qs, cs);
                swizB(stream, cs, csw, 256, MSGD, 0, MSGD, (long long)MPAD * MSGD);
                swizB(stream, qs, qsw, 256, MSGD, 0, MSGD, (long long)MPAD * MSGD);
                // q_int = plan @ cs  -> inter rows [b*432 .. +EQ)
                gemm_bat<<<dim3(2, MSGD / 32, BB), 256, 0, stream>>>(
                    plan, 256, 65536LL, csw, 256LL * MSGD,
                    inter, 1, MSGD, (long long)EPG * MSGD, MSGD, 256, 1.f, EQv);
                // c_int = planT @ qs -> inter rows [b*432+EQ .. +EC)
                gemm_bat<<<dim3(2, MSGD / 32, BB), 256, 0, stream>>>(
                    planT, 256, 65536LL, qsw, 256LL * MSGD,
                    inter + (size_t)EQv * MSGD, 1, MSGD, (long long)EPG * MSGD,
                    MSGD, 256, 1.f, ECv);
            }
        }

        // ---- sinkhorn features + plan ----
        split_pad_k<<<(unsigned)splitGrid, 256, 0, stream>>>(e, qs, cs);
        gemm(stream, qs, MSGD, nullptr, nullptr, 0, nullptr, nullptr, 0, nullptr,
             wS1, sb1, hid, nullptr, BB * MPAD, SDIMv, 1, 0);
        gemm(stream, hid, SDIMv, nullptr, nullptr, 0, nullptr, nullptr, 0, nullptr,
             wS2, sb2, fq, nullptr, BB * MPAD, SDIMv, 0, 0, MPAD, EQv);
        gemm(stream, cs, MSGD, nullptr, nullptr, 0, nullptr, nullptr, 0, nullptr,
             wS1, sb1, hid, nullptr, BB * MPAD, SDIMv, 1, 0);
        gemm(stream, hid, SDIMv, nullptr, nullptr, 0, nullptr, nullptr, 0, nullptr,
             wS2, sb2, fc, nullptr, BB * MPAD, SDIMv, 0, 0, MPAD, ECv);

        // la = (fq @ fc^T) / TEMP  (alpha=10); B = fc^T swizzled (trans read)
        swizB(stream, fc, fsw, SDIMv, MPAD, 1, SDIMv, (long long)MPAD * SDIMv);
        gemm_bat<<<dim3(2, MPAD / 32, BB), 256, 0, stream>>>(
            fq, SDIMv, (long long)MPAD * SDIMv, fsw, (long long)SDIMv * MPAD,
            la, 0, MPAD, 65536LL, MPAD, SDIMv, 10.f, MPAD);

        for (int it = 0; it < 10; ++it) {
            sink_rownorm_k<<<BB * MPAD, 256, 0, stream>>>(la);
            sink_colnorm_k<<<BB * MPAD, 256, 0, stream>>>(la, plan, planT,
                                                          it == 9 ? 1 : 0);
        }
    }

    // ---- alignment score: out[b] = -|| qs - plan @ cs ||_1 ----
    swizB(stream, cs, csw, 256, MSGD, 0, MSGD, (long long)MPAD * MSGD);
    gemm_bat<<<dim3(2, MSGD / 32, BB), 256, 0, stream>>>(
        plan, 256, 65536LL, csw, 256LL * MSGD,
        tb, 0, MSGD, (long long)MPAD * MSGD, MSGD, 256, 1.f, MPAD);
    score_k<<<BB, 256, 0, stream>>>(qs, tb, (float*)d_out);
}
